// SRULayer_9552007266657
// MI455X (gfx1250) — matrix operand order repaired and checked by reading, not compiled, not run on hardware
//
#include <hip/hip_runtime.h>

// ---------------------------------------------------------------------------
// SRU layer for MI455X (gfx1250, wave32, WMMA).
//   x:(2048,8,1024) f32 -> LN -> GEMM(16384x1024 @ 1024x3072, split-bf16 WMMA)
//   -> blocked affine scan over L -> fused epilogue.
// GEMM: 32(M) x 64(N) tile per wave; K loop = outer(256) x unrolled inner(32).
// All loads are indexed off the kernarg pointers with integer offsets so the
// backend keeps them on the GLOBAL path (single LOADcnt, no FLAT/LDS-pipe
// arbitration); inner offsets are compile-time immediates.
// ---------------------------------------------------------------------------

typedef __attribute__((ext_vector_type(8)))  __bf16 v8bf;
typedef __attribute__((ext_vector_type(16))) __bf16 v16bf;
typedef __attribute__((ext_vector_type(8)))  float  v8f;

#define SEQ_L 2048
#define BATCH 8
#define DIM   1024
#define E3    3072               // 3*DIM
#define ROWS  (SEQ_L * BATCH)    // 16384
#define NCHUNK 16
#define CHUNK_LEN (SEQ_L / NCHUNK)  // 128
#define NCHAN (BATCH * DIM)         // 8192

// ------------------------------ LayerNorm ----------------------------------
__global__ __launch_bounds__(256)
void sru_ln_kernel(const float* __restrict__ x,
                   const float* __restrict__ gamma,
                   const float* __restrict__ beta,
                   __bf16* __restrict__ xn_hi,
                   __bf16* __restrict__ xn_lo) {
    const int row = blockIdx.x;
    const float* xr = x + (size_t)row * DIM;

    __shared__ float s_sum[256];
    __shared__ float s_sq[256];
    float sum = 0.0f, sq = 0.0f;
    for (int i = threadIdx.x; i < DIM; i += 256) {
        float v = xr[i];
        sum += v;
        sq  += v * v;
    }
    s_sum[threadIdx.x] = sum;
    s_sq[threadIdx.x]  = sq;
    __syncthreads();
    for (int s = 128; s > 0; s >>= 1) {
        if (threadIdx.x < (unsigned)s) {
            s_sum[threadIdx.x] += s_sum[threadIdx.x + s];
            s_sq[threadIdx.x]  += s_sq[threadIdx.x + s];
        }
        __syncthreads();
    }
    const float mu  = s_sum[0] * (1.0f / DIM);
    const float var = s_sq[0] * (1.0f / DIM) - mu * mu;
    const float inv = rsqrtf(var + 1e-5f);

    for (int i = threadIdx.x; i < DIM; i += 256) {
        float xv = (xr[i] - mu) * inv * gamma[i] + beta[i];
        __bf16 h = (__bf16)xv;
        __bf16 l = (__bf16)(xv - (float)h);
        xn_hi[(size_t)row * DIM + i] = h;
        xn_lo[(size_t)row * DIM + i] = l;
    }
}

// --------------------- W split + transpose to [N][K] -----------------------
__global__ __launch_bounds__(256)
void sru_wsplit_kernel(const float* __restrict__ W,
                       __bf16* __restrict__ wt_hi,
                       __bf16* __restrict__ wt_lo) {
    int idx = blockIdx.x * 256 + threadIdx.x;   // 0 .. DIM*E3-1
    int k = idx & (DIM - 1);
    int n = idx >> 10;
    float w = W[(size_t)k * E3 + n];
    __bf16 h = (__bf16)w;
    wt_hi[(size_t)n * DIM + k] = h;
    wt_lo[(size_t)n * DIM + k] = (__bf16)(w - (float)h);
}

// ------------------------------- GEMM --------------------------------------
// ufr = xn @ W + b  via V_WMMA_F32_16X16X32_BF16, split-bf16 3-term emulation:
// (Ah+Al)(Bh+Bl) ~= Ah*Bh + Ah*Bl + Al*Bh  (dropped Al*Bl term is ~2^-16).
// Block = 128 threads = 4 waves; each wave owns a 32(M) x 64(N) output tile.
#define SRU_TILE(BOFF, T)                                                      \
    do {                                                                       \
        v16bf Bh_ = *(const v16bf*)(wt_hi + (BOFF) + kk);                      \
        v16bf Bl_ = *(const v16bf*)(wt_lo + (BOFF) + kk);                      \
        acc0[T] = __builtin_amdgcn_wmma_f32_16x16x32_bf16(                     \
            false, Ah0, false, Bh_, (short)0, acc0[T], false, false);          \
        acc0[T] = __builtin_amdgcn_wmma_f32_16x16x32_bf16(                     \
            false, Ah0, false, Bl_, (short)0, acc0[T], false, false);          \
        acc0[T] = __builtin_amdgcn_wmma_f32_16x16x32_bf16(                     \
            false, Al0, false, Bh_, (short)0, acc0[T], false, false);          \
        acc1[T] = __builtin_amdgcn_wmma_f32_16x16x32_bf16(                     \
            false, Ah1, false, Bh_, (short)0, acc1[T], false, false);          \
        acc1[T] = __builtin_amdgcn_wmma_f32_16x16x32_bf16(                     \
            false, Ah1, false, Bl_, (short)0, acc1[T], false, false);          \
        acc1[T] = __builtin_amdgcn_wmma_f32_16x16x32_bf16(                     \
            false, Al1, false, Bh_, (short)0, acc1[T], false, false);          \
    } while (0)

__global__ __launch_bounds__(128)
void sru_gemm_kernel(const __bf16* __restrict__ xn_hi,
                     const __bf16* __restrict__ xn_lo,
                     const __bf16* __restrict__ wt_hi,
                     const __bf16* __restrict__ wt_lo,
                     const float*  __restrict__ bias,
                     float* __restrict__ ufr) {
    const int lane  = threadIdx.x & 31;
    const int wave  = threadIdx.x >> 5;
    const int half  = lane >> 4;     // 0: lanes 0-15, 1: lanes 16-31
    const int l16   = lane & 15;
    const int mBase = (blockIdx.x * 8 + wave * 2) * 16;   // two M tiles per wave
    const int nBase = blockIdx.y * 64;

    v8f acc0[4], acc1[4];
    #pragma unroll
    for (int t = 0; t < 4; ++t) { acc0[t] = (v8f){}; acc1[t] = (v8f){}; }

    // Integer offsets off the kernarg pointers (addrspace(1) provable).
    // A fragment (16x32 bf16): VGPR0-3 = K[kb+half*16 .. +7],
    //                          VGPR4-7 = K[kb+half*16+8 .. +7]
    const size_t a0 = (size_t)(mBase + l16) * DIM + half * 16;
    const size_t a1 = a0 + (size_t)16 * DIM;
    // B fragment (32x16 bf16): lane holds col n = nBase+t*16+l16,
    // 16 contiguous K starting at kb+half*16.
    const size_t b0 = (size_t)(nBase + l16) * DIM + half * 16;
    const size_t b1 = b0 + (size_t)16 * DIM;
    const size_t b2 = b0 + (size_t)32 * DIM;
    const size_t b3 = b0 + (size_t)48 * DIM;

    for (int ko = 0; ko < DIM; ko += 256) {
        // Prefetch the next 256-K panel of A (global_prefetch_b8).
        __builtin_prefetch(xn_hi + a0 + ko + 256, 0, 1);
        __builtin_prefetch(xn_lo + a0 + ko + 256, 0, 1);
        __builtin_prefetch(xn_hi + a1 + ko + 256, 0, 1);
        __builtin_prefetch(xn_lo + a1 + ko + 256, 0, 1);

        #pragma unroll
        for (int ks = 0; ks < 256; ks += 32) {
            const int kk = ko + ks;
            v8bf a0h0 = *(const v8bf*)(xn_hi + a0 + kk);
            v8bf a1h0 = *(const v8bf*)(xn_hi + a0 + kk + 8);
            v8bf a0l0 = *(const v8bf*)(xn_lo + a0 + kk);
            v8bf a1l0 = *(const v8bf*)(xn_lo + a0 + kk + 8);
            v8bf a0h1 = *(const v8bf*)(xn_hi + a1 + kk);
            v8bf a1h1 = *(const v8bf*)(xn_hi + a1 + kk + 8);
            v8bf a0l1 = *(const v8bf*)(xn_lo + a1 + kk);
            v8bf a1l1 = *(const v8bf*)(xn_lo + a1 + kk + 8);
            v16bf Ah0, Al0, Ah1, Al1;
            #pragma unroll
            for (int i = 0; i < 8; ++i) {
                Ah0[i] = a0h0[i]; Ah0[i + 8] = a1h0[i];
                Al0[i] = a0l0[i]; Al0[i + 8] = a1l0[i];
                Ah1[i] = a0h1[i]; Ah1[i + 8] = a1h1[i];
                Al1[i] = a0l1[i]; Al1[i + 8] = a1l1[i];
            }

            SRU_TILE(b0, 0);
            SRU_TILE(b1, 1);
            SRU_TILE(b2, 2);
            SRU_TILE(b3, 3);
        }
    }

    // Epilogue: fuse bias add. C/D layout: VGPR i -> row base+i+8*half,
    // col nBase+t*16+l16.
    #pragma unroll
    for (int t = 0; t < 4; ++t) {
        const int col = nBase + t * 16 + l16;
        const float bv = bias[col];
        #pragma unroll
        for (int i = 0; i < 8; ++i) {
            ufr[(size_t)(mBase + i + half * 8) * E3 + col]      = acc0[t][i] + bv;
            ufr[(size_t)(mBase + 16 + i + half * 8) * E3 + col] = acc1[t][i] + bv;
        }
    }
}

__device__ __forceinline__ float sigmoidf_(float v) {
    return 1.0f / (1.0f + expf(-v));
}

// ------------------- Scan phase A: per-chunk affine transform --------------
__global__ __launch_bounds__(256)
void sru_scan_a_kernel(const float* __restrict__ ufr,
                       float* __restrict__ chA,
                       float* __restrict__ chB) {
    int tid   = blockIdx.x * 256 + threadIdx.x;   // 0 .. NCHAN*NCHUNK-1
    int ch    = tid & (NCHAN - 1);
    int chunk = tid >> 13;                        // NCHAN = 2^13
    int b = ch >> 10;
    int d = ch & (DIM - 1);
    float A = 1.0f, Bc = 0.0f;
    for (int l = chunk * CHUNK_LEN; l < (chunk + 1) * CHUNK_LEN; ++l) {
        size_t base = (size_t)(l * BATCH + b) * E3 + d;
        float u = ufr[base];
        float f = sigmoidf_(ufr[base + DIM]);
        Bc = f * Bc + (1.0f - f) * u;
        A *= f;
    }
    chA[tid] = A;
    chB[tid] = Bc;
}

// ------------------- Scan phase B: combine chunks sequentially -------------
__global__ __launch_bounds__(256)
void sru_scan_b_kernel(const float* __restrict__ c0,
                       const float* __restrict__ chA,
                       const float* __restrict__ chB,
                       float* __restrict__ cin,
                       float* __restrict__ last_c) {
    int ch = blockIdx.x * 256 + threadIdx.x;      // 0 .. NCHAN-1
    float c = c0[ch];
    for (int k = 0; k < NCHUNK; ++k) {
        cin[k * NCHAN + ch] = c;
        c = chA[k * NCHAN + ch] * c + chB[k * NCHAN + ch];
    }
    last_c[ch] = c;
}

// ------------------- Scan phase C: materialize + fused epilogue ------------
__global__ __launch_bounds__(256)
void sru_scan_c_kernel(const float* __restrict__ x,
                       const float* __restrict__ ufr,
                       const __bf16* __restrict__ xn_hi,
                       const __bf16* __restrict__ xn_lo,
                       const float* __restrict__ cin,
                       float* __restrict__ out) {
    int tid   = blockIdx.x * 256 + threadIdx.x;
    int ch    = tid & (NCHAN - 1);
    int chunk = tid >> 13;
    int b = ch >> 10;
    int d = ch & (DIM - 1);
    float c = cin[chunk * NCHAN + ch];
    for (int l = chunk * CHUNK_LEN; l < (chunk + 1) * CHUNK_LEN; ++l) {
        size_t base3 = (size_t)(l * BATCH + b) * E3 + d;
        size_t base1 = (size_t)(l * BATCH + b) * DIM + d;
        float u = ufr[base3];
        float f = sigmoidf_(ufr[base3 + DIM]);
        float r = sigmoidf_(ufr[base3 + 2 * DIM]);
        c = f * c + (1.0f - f) * u;
        float xn = (float)xn_hi[base1] + (float)xn_lo[base1];
        float hs = r * tanhf(c) + (1.0f - r) * xn;
        out[base1] = x[base1] + hs;
    }
}

// ------------------------------ launcher -----------------------------------
extern "C" void kernel_launch(void* const* d_in, const int* in_sizes, int n_in,
                              void* d_out, int out_size, void* d_ws, size_t ws_size,
                              hipStream_t stream) {
    const float* x     = (const float*)d_in[0];  // (2048, 8, 1024)
    const float* c0    = (const float*)d_in[1];  // (8, 1024)
    const float* W     = (const float*)d_in[2];  // (1024, 3072)
    const float* bias  = (const float*)d_in[3];  // (3072,)
    const float* gamma = (const float*)d_in[4];  // (1024,)
    const float* beta  = (const float*)d_in[5];  // (1024,)
    float* out = (float*)d_out;                  // out1 (L*B*D) ++ last_c (B*D)

    // Workspace layout (all offsets 64B-aligned).
    char* w = (char*)d_ws;
    __bf16* xn_hi = (__bf16*)(w);                           //  33,554,432 B
    __bf16* xn_lo = (__bf16*)(w + 33554432ull);             //  33,554,432 B
    __bf16* wt_hi = (__bf16*)(w + 67108864ull);             //   6,291,456 B
    __bf16* wt_lo = (__bf16*)(w + 73400320ull);             //   6,291,456 B
    float*  ufr   = (float*) (w + 79691776ull);             // 201,326,592 B
    float*  chA   = (float*) (w + 281018368ull);            //     524,288 B
    float*  chB   = (float*) (w + 281542656ull);            //     524,288 B
    float*  cin   = (float*) (w + 282066944ull);            //     524,288 B

    // 1) LayerNorm + bf16 hi/lo split of x_norm
    sru_ln_kernel<<<ROWS, 256, 0, stream>>>(x, gamma, beta, xn_hi, xn_lo);

    // 2) W split + transpose to [N][K]
    sru_wsplit_kernel<<<(DIM * E3) / 256, 256, 0, stream>>>(W, wt_hi, wt_lo);

    // 3) GEMM + bias (WMMA bf16 split path), 32x64 tile per wave
    dim3 ggrid(ROWS / 16 / 8, E3 / 64);   // (128, 48)
    sru_gemm_kernel<<<ggrid, 128, 0, stream>>>(xn_hi, xn_lo, wt_hi, wt_lo, bias, ufr);

    // 4) Blocked scan: per-chunk affine transforms
    sru_scan_a_kernel<<<(NCHAN * NCHUNK) / 256, 256, 0, stream>>>(ufr, chA, chB);

    // 5) Combine chunk transforms; emits last_c
    sru_scan_b_kernel<<<NCHAN / 256, 256, 0, stream>>>(c0, chA, chB, cin,
                                                       out + (size_t)ROWS * DIM);

    // 6) Materialize c, fused tanh/r-gate/x_norm blend + residual
    sru_scan_c_kernel<<<(NCHAN * NCHUNK) / 256, 256, 0, stream>>>(x, ufr, xn_hi,
                                                                  xn_lo, cin, out);
}
